// BahdanauAttention_68539088109695
// MI455X (gfx1250) — matrix operand order repaired and checked by reading, not compiled, not run on hardware
//
#include <hip/hip_runtime.h>
#include <hip/hip_bf16.h>

// ---------------------------------------------------------------------------
// Bahdanau attention, MI455X (gfx1250, wave32, WMMA bf16 16x16x32).
//   B=32, T=2048, H=1024, K=2H=2048 (A = [query || key], W = [Wq || Wk])
// Pipeline:
//   0) prep_w     : convert [Wq||Wk] fp32 -> bf16 once into d_ws (4 MB)
//   1) score_gemm : fused (q+k) projection GEMM (bf16 WMMA, f32 acc),
//                   double-buffered W chunks staged with async global->LDS,
//                   + bias + tanh + dot(Wz) + mask -> scores into d_out
//   2) softmax_t  : per-batch masked softmax in-place
//   3) zero_out + wvalue : att_value = weights^T @ value (atomic f32 accumulate)
// ---------------------------------------------------------------------------

typedef __attribute__((ext_vector_type(16))) __bf16 v16bf;
typedef __attribute__((ext_vector_type(8)))  __bf16 v8bf;
typedef __attribute__((ext_vector_type(8)))  float  v8f;
typedef int v4i __attribute__((vector_size(16)));   // matches async-LDS builtin param
typedef unsigned int v2u __attribute__((vector_size(8)));

// Explicit LDS (addrspace 3) types: guarantees ds_load/ds_store codegen and
// avoids per-access generic->AS3 null-check selects.
typedef __attribute__((address_space(3))) unsigned short lus;
typedef __attribute__((address_space(3))) v8bf  a3v8bf;
typedef __attribute__((address_space(3))) v4i   a3v4i;
typedef __attribute__((address_space(3))) v2u   a3v2u;
typedef __attribute__((address_space(1))) v4i   a1v4i;

#define BATCH   32
#define T_LEN   2048
#define H_DIM   1024
#define K_DIM   2048     // 2*H (concat query/key feature dim)
#define M_TILE  64       // rows (b,t) per workgroup
#define N_TILE  128      // output features per pass
#define KC      64       // K per staged W chunk (2 WMMA k-steps)
#define NK      (K_DIM / KC)   // 32 chunks
#define A_RS    2064     // A row stride (bf16): 2048 + 16 pad -> 8-bank lane stride
#define W_RS    (KC + 8) // 72: W chunk row stride (bf16); 144B row = 16B aligned

#if defined(__gfx1250__) && __has_builtin(__builtin_amdgcn_global_load_async_to_lds_b128)
#define USE_ASYNC_LDS 1
#else
#define USE_ASYNC_LDS 0
#endif

#if USE_ASYNC_LDS
#if __has_builtin(__builtin_amdgcn_s_wait_asynccnt)
#define WAIT_ASYNC() __builtin_amdgcn_s_wait_asynccnt(0)
#else
#define WAIT_ASYNC() asm volatile("s_wait_asynccnt 0x0" ::: "memory")
#endif
#else
#define WAIT_ASYNC()
#endif

__device__ __forceinline__ unsigned int f32_bf16(float f) {
  unsigned int u = __float_as_uint(f);
  u += 0x7fffu + ((u >> 16) & 1u);           // round-to-nearest-even
  return u >> 16;
}

// Copy one [N_TILE x KC] bf16 chunk of the pre-converted W into LDS.
// Pure 16-byte copies; async global->LDS (ASYNCcnt) when available.
__device__ __forceinline__ void stage_w(lus* dst, const unsigned short* src_nn,
                                        int kk, int tid) {
  #pragma unroll
  for (int q = 0; q < 4; ++q) {
    int i  = (q << 8) + tid;        // 0..1023 16-byte transfers
    int r  = i >> 3;                // 8 per row
    int c8 = (i & 7) << 3;          // bf16 col within chunk
    const unsigned short* g = src_nn + (size_t)r * K_DIM + kk + c8;
    lus* l = dst + r * W_RS + c8;
#if USE_ASYNC_LDS
    __builtin_amdgcn_global_load_async_to_lds_b128(
        (a1v4i*)(void*)g, (a3v4i*)l, 0, 0);
#else
    *(a3v4i*)l = *(const v4i*)g;
#endif
  }
}

// One KC=64 chunk of WMMA work: 2 k-substeps x 4 n-subtiles = 8 WMMAs.
// All fragments for a substep are preloaded into distinct registers first so
// the ds_load_b128s can be clustered and drained with partial dscnt waits
// while the WMMA chain issues (avoids load->wait0->wmma lockstep).
__device__ __forceinline__ void compute_chunk(const lus* wbuf, const lus* abase,
                                              int kk, int nrowbase,
                                              int koffA, int koffB, v8f acc[4]) {
  #pragma unroll
  for (int s = 0; s < 2; ++s) {
    const lus* ap = abase + kk + (s << 5) + koffA;
    v8bf a0 = *(const a3v8bf*)ap;
    v8bf a1 = *(const a3v8bf*)(ap + 16);
    v8bf b0[4], b1[4];
    #pragma unroll
    for (int j = 0; j < 4; ++j) {
      const lus* bp = wbuf + (nrowbase + j * 16) * W_RS + (s << 5) + koffB;
      b0[j] = *(const a3v8bf*)bp;
      b1[j] = *(const a3v8bf*)(bp + 16);
    }
    v16bf afrag;
    #pragma unroll
    for (int e = 0; e < 8; ++e) { afrag[e] = a0[e]; afrag[e + 8] = a1[e]; }
    #pragma unroll
    for (int j = 0; j < 4; ++j) {
      v16bf bfrag;
      #pragma unroll
      for (int e = 0; e < 8; ++e) { bfrag[e] = b0[j][e]; bfrag[e + 8] = b1[j][e]; }
      acc[j] = __builtin_amdgcn_wmma_f32_16x16x32_bf16(
          false, afrag, false, bfrag, (short)0, acc[j], false, false);
    }
  }
}

// ---------------------------------------------------------------------------
// Kernel 0: W prepass, [Wq||Wk] fp32 -> bf16 [1024 x 2048] row-major in d_ws.
// ---------------------------------------------------------------------------
__global__ void __launch_bounds__(256)
prep_w(const float* __restrict__ Wq, const float* __restrict__ Wk,
       unsigned short* __restrict__ out)
{
  int i = blockIdx.x * 256 + threadIdx.x;   // over float4s: 2M/4 = 524288
  int r = i >> 9;                           // 512 float4 per row
  int c = (i & 511) << 2;
  const float* src = (c < H_DIM) ? (Wq + (size_t)r * H_DIM + c)
                                 : (Wk + (size_t)r * H_DIM + (c - H_DIM));
  float4 v = *(const float4*)src;
  v2u pk = { f32_bf16(v.x) | (f32_bf16(v.y) << 16),
             f32_bf16(v.z) | (f32_bf16(v.w) << 16) };
  *(v2u*)(out + (size_t)r * K_DIM + c) = pk;
}

// ---------------------------------------------------------------------------
// Kernel 1: fused projection + score.
// grid = (B*T)/M_TILE = 1024 blocks, 256 threads (8 waves).
// ---------------------------------------------------------------------------
__global__ void __launch_bounds__(256, 1)
score_gemm(const float* __restrict__ query, const float* __restrict__ key_,
           const unsigned short* __restrict__ wsW,   // bf16 [1024 x 2048]
           const float* __restrict__ bq,    const float* __restrict__ bk,
           const float* __restrict__ Wz,    const float* __restrict__ bz,
           const int*   __restrict__ lengths,
           float* __restrict__ scores)
{
  extern __shared__ char smem[];
  lus* Abuf = (lus*)(__attribute__((address_space(3))) char*)smem;   // 64 x 2064 bf16
  lus* Wb0  = Abuf + M_TILE * A_RS;                                  // 128 x 72 bf16
  lus* Wb1  = Wb0 + N_TILE * W_RS;
  float* scoreAcc = (float*)(smem + M_TILE * A_RS * 2 + 2 * N_TILE * W_RS * 2);

  const int tid   = threadIdx.x;
  const int lane  = tid & 31;
  const int wv    = tid >> 5;
  const int m_sub = wv & 3;             // 4 m-subtiles of 16 rows
  const int ngrp  = wv >> 2;            // 2 n-groups of 64 cols
  const int cl    = lane & 15;
  const int koffA = (lane >> 4) << 3;   // A frag K split {0-7,16-23}/{8-15,24-31}
  const int koffB = (lane >> 4) << 3;   // B frag K split {0-7,16-23}/{8-15,24-31}
  const int nrowbase = ngrp * 64 + cl;  // wave-constant B row base in W chunk
  const long long row0 = (long long)blockIdx.x * M_TILE;

  if (tid < M_TILE) scoreAcc[tid] = 0.f;

  // Stage A tile [64 x 2048] = [query || key], fp32 -> bf16, into LDS (once).
  for (int i = tid; i < M_TILE * (H_DIM / 4); i += 256) {
    int r = i >> 8;                     // 256 float4 per row per source
    int k = (i & 255) << 2;
    float4 v = *(const float4*)(query + (row0 + r) * H_DIM + k);
    v2u pk = { f32_bf16(v.x) | (f32_bf16(v.y) << 16),
               f32_bf16(v.z) | (f32_bf16(v.w) << 16) };
    *(a3v2u*)(Abuf + r * A_RS + k) = pk;
  }
  for (int i = tid; i < M_TILE * (H_DIM / 4); i += 256) {
    int r = i >> 8;
    int k = (i & 255) << 2;
    float4 v = *(const float4*)(key_ + (row0 + r) * H_DIM + k);
    v2u pk = { f32_bf16(v.x) | (f32_bf16(v.y) << 16),
               f32_bf16(v.z) | (f32_bf16(v.w) << 16) };
    *(a3v2u*)(Abuf + r * A_RS + H_DIM + k) = pk;
  }

  const lus* abase = Abuf + (m_sub * 16 + cl) * A_RS;

  for (int nn = 0; nn < H_DIM; nn += N_TILE) {
    const unsigned short* wsnn = wsW + (size_t)nn * K_DIM;

    v8f acc[4];
    #pragma unroll
    for (int j = 0; j < 4; ++j)
      #pragma unroll
      for (int e = 0; e < 8; ++e) acc[j][e] = 0.f;

    stage_w(Wb0, wsnn, 0, tid);         // prologue chunk

    // Manually double-stepped so both staging and compute use constant LDS
    // pointers (keeps everything on the ds_ path, no flat fallback).
    for (int kki = 0; kki < NK; kki += 2) {
      WAIT_ASYNC();
      __syncthreads();                  // Wb0 staged; prior Wb0 reads done
      if (kki + 1 < NK) stage_w(Wb1, wsnn, (kki + 1) * KC, tid);
      compute_chunk(Wb0, abase, kki * KC, nrowbase, koffA, koffB, acc);

      WAIT_ASYNC();
      __syncthreads();                  // Wb1 staged; prior Wb1 reads done
      if (kki + 2 < NK) stage_w(Wb0, wsnn, (kki + 2) * KC, tid);
      compute_chunk(Wb1, abase, (kki + 1) * KC, nrowbase, koffA, koffB, acc);
    }

    // Fused epilogue: bias + tanh + Wz, reduce the 16 columns per lane-half.
    // C layout: lane<16 holds rows m_sub*16+g, lane>=16 rows m_sub*16+8+g, col=cl.
    float part[8];
    #pragma unroll
    for (int g = 0; g < 8; ++g) part[g] = 0.f;
    #pragma unroll
    for (int j = 0; j < 4; ++j) {
      int col = nn + ngrp * 64 + j * 16 + cl;
      float bias = bq[col] + bk[col];
      float wz   = Wz[col];
      #pragma unroll
      for (int g = 0; g < 8; ++g)
        part[g] += tanhf(acc[j][g] + bias) * wz;
    }
    #pragma unroll
    for (int g = 0; g < 8; ++g) {       // xor-butterfly stays inside 16-lane half
      float p = part[g];
      p += __shfl_xor(p, 1); p += __shfl_xor(p, 2);
      p += __shfl_xor(p, 4); p += __shfl_xor(p, 8);
      part[g] = p;
    }
    if (cl == 0) {
      int rbase = m_sub * 16 + ((lane >> 4) << 3);
      #pragma unroll
      for (int g = 0; g < 8; ++g) atomicAdd(&scoreAcc[rbase + g], part[g]);
    }
  }

  __syncthreads();
  if (tid < M_TILE) {
    long long gt = row0 + tid;
    int b = (int)(gt >> 11);
    int t = (int)(gt & (T_LEN - 1));
    float sc = scoreAcc[tid] + bz[0];
    if (t >= lengths[b]) sc = -1e9f;    // reference NEG mask
    scores[gt] = sc;
  }
}

// ---------------------------------------------------------------------------
// Kernel 2: per-batch softmax over T, in-place on the scores region.
// ---------------------------------------------------------------------------
__global__ void __launch_bounds__(256)
softmax_t(float* __restrict__ wts)
{
  __shared__ float red[256];
  const int tid = threadIdx.x;
  float* p = wts + (long long)blockIdx.x * T_LEN;
  float v[8], m = -3.402823466e38f;
  #pragma unroll
  for (int i = 0; i < 8; ++i) { v[i] = p[tid + i * 256]; m = fmaxf(m, v[i]); }
  red[tid] = m; __syncthreads();
  for (int s = 128; s > 0; s >>= 1) {
    if (tid < s) red[tid] = fmaxf(red[tid], red[tid + s]);
    __syncthreads();
  }
  m = red[0]; __syncthreads();
  float sum = 0.f;
  #pragma unroll
  for (int i = 0; i < 8; ++i) { v[i] = __expf(v[i] - m); sum += v[i]; }
  red[tid] = sum; __syncthreads();
  for (int s = 128; s > 0; s >>= 1) {
    if (tid < s) red[tid] += red[tid + s];
    __syncthreads();
  }
  float inv = 1.f / red[0];
  #pragma unroll
  for (int i = 0; i < 8; ++i) p[tid + i * 256] = v[i] * inv;
}

// ---------------------------------------------------------------------------
// Kernel 3: att_value[b,:] = sum_t w[b,t] * value[b,t,:]  (memory bound)
// ---------------------------------------------------------------------------
__global__ void zero_out(float* __restrict__ p, int n) {
  int i = blockIdx.x * 256 + threadIdx.x;
  if (i < n) p[i] = 0.f;
}

__global__ void __launch_bounds__(256)
wvalue(const float* __restrict__ value, const float* __restrict__ wts,
       float* __restrict__ out)
{
  __shared__ float wbuf[256];
  const int tid   = threadIdx.x;
  const int b     = blockIdx.x >> 3;
  const int slice = blockIdx.x & 7;           // 8 slices of 256 t each
  const long long tbase = (long long)b * T_LEN + slice * 256;
  wbuf[tid] = wts[tbase + tid];
  __syncthreads();
  float4 acc = {0.f, 0.f, 0.f, 0.f};
  const float4* vp = (const float4*)(value + tbase * H_DIM);
  for (int tt = 0; tt < 256; ++tt) {
    float w = wbuf[tt];
    float4 x = vp[(long long)tt * (H_DIM / 4) + tid];
    acc.x += w * x.x; acc.y += w * x.y; acc.z += w * x.z; acc.w += w * x.w;
  }
  float* o = out + (long long)b * H_DIM + tid * 4;
  unsafeAtomicAdd(o + 0, acc.x);
  unsafeAtomicAdd(o + 1, acc.y);
  unsafeAtomicAdd(o + 2, acc.z);
  unsafeAtomicAdd(o + 3, acc.w);
}

// ---------------------------------------------------------------------------
extern "C" void kernel_launch(void* const* d_in, const int* in_sizes, int n_in,
                              void* d_out, int out_size, void* d_ws, size_t ws_size,
                              hipStream_t stream) {
  (void)in_sizes; (void)n_in; (void)out_size; (void)ws_size;
  const float* query   = (const float*)d_in[0];
  const float* key_    = (const float*)d_in[1];
  const float* value   = (const float*)d_in[2];
  const int*   lengths = (const int*)  d_in[3];
  const float* Wq      = (const float*)d_in[4];
  const float* bq      = (const float*)d_in[5];
  const float* Wk      = (const float*)d_in[6];
  const float* bk      = (const float*)d_in[7];
  const float* Wz      = (const float*)d_in[8];
  const float* bz      = (const float*)d_in[9];

  float* att_value = (float*)d_out;                      // [B,1,H] = 32768 floats
  float* att_w     = (float*)d_out + BATCH * H_DIM;      // [B,T,1] = 65536 floats
  unsigned short* wsW = (unsigned short*)d_ws;           // bf16 [1024 x 2048], 4 MB

  const size_t smem = (size_t)M_TILE * A_RS * 2 +        // A tile   (~258 KB)
                      (size_t)2 * N_TILE * W_RS * 2 +    // W double buffer (~36 KB)
                      (size_t)M_TILE * sizeof(float);    // score accumulators
  const int gemm_blocks = (BATCH * T_LEN) / M_TILE;      // 1024

  prep_w<<<(H_DIM * K_DIM / 4) / 256, 256, 0, stream>>>(Wq, Wk, wsW);
  score_gemm<<<gemm_blocks, 256, smem, stream>>>(
      query, key_, wsW, bq, bk, Wz, bz, lengths, att_w);
  softmax_t<<<BATCH, 256, 0, stream>>>(att_w);
  zero_out<<<(BATCH * H_DIM) / 256, 256, 0, stream>>>(att_value, BATCH * H_DIM);
  wvalue<<<BATCH * 8, 256, 0, stream>>>(value, att_w, att_value);
}
